// MultiHead_34952443854972
// MI455X (gfx1250) — matrix operand order repaired and checked by reading, not compiled, not run on hardware
//
#include <hip/hip_runtime.h>
#include <math.h>

// ---------------------------------------------------------------------------
// Types for CDNA5 WMMA / TDM
// ---------------------------------------------------------------------------
typedef __attribute__((ext_vector_type(16))) __bf16        v16bf;
typedef __attribute__((ext_vector_type(8)))  float         v8f;
typedef __attribute__((ext_vector_type(8)))  unsigned int  v8u;
typedef __attribute__((ext_vector_type(4)))  unsigned int  v4u;
typedef __attribute__((ext_vector_type(8)))  int           v8i;
typedef __attribute__((ext_vector_type(4)))  int           v4i;

__device__ __forceinline__ v16bf as_v16bf(v8u u) {
    union { v8u u; v16bf b; } c; c.u = u; return c.b;
}

// round-to-nearest-even f32 -> bf16
__device__ __forceinline__ unsigned int pack_bf16(float a, float b) {
    unsigned int ua = __builtin_bit_cast(unsigned int, a);
    unsigned int ub = __builtin_bit_cast(unsigned int, b);
    ua += 0x7FFFu + ((ua >> 16) & 1u);
    ub += 0x7FFFu + ((ub >> 16) & 1u);
    return (ua >> 16) | (ub & 0xFFFF0000u);
}
__device__ __forceinline__ unsigned short bf16_of(float a) {
    unsigned int ua = __builtin_bit_cast(unsigned int, a);
    ua += 0x7FFFu + ((ua >> 16) & 1u);
    return (unsigned short)(ua >> 16);
}

#define WMMA_BF16(A, B, C) \
    __builtin_amdgcn_wmma_f32_16x16x32_bf16(false, (A), false, (B), (short)0, (C), false, false)

// ---------------------------------------------------------------------------
// Tensor Data Mover: async 2D bf16 tile load global -> LDS, with LDS row pad.
// D# per CDNA5 ISA 8.3/8.4. pad_interval code: pad after (1<<code)*8 bytes;
// pad_amount code 0 => 1 dword inserted.
// ---------------------------------------------------------------------------
#if __has_include(<hip/amd_detail/amd_gfx1250_TDM.h>)
#define TDM_6ARG 1
#endif

__device__ __forceinline__ void tdm_load_2d_bf16(
    unsigned int lds_addr, const void* gaddr,
    unsigned int tensor_d0, unsigned int tensor_d1,
    unsigned int tile_d0, unsigned int tile_d1,
    unsigned long long stride_elems, unsigned int pad_code)
{
    unsigned long long ga = (unsigned long long)gaddr;
    v4u g0;
    g0[0] = 1u;                                   // count=1, user descriptor
    g0[1] = lds_addr;                             // LDS byte address
    g0[2] = (unsigned int)ga;                     // global addr [31:0]
    g0[3] = (unsigned int)((ga >> 32) & 0x01FFFFFFull) | 0x80000000u; // [56:32] | type=2
    v8i g1;
    g1[0] = (int)((1u << 16) | (1u << 20) | (pad_code << 22));  // data_size=2B, pad_enable, pad_interval, pad_amount=0(1dw)
    g1[1] = (int)((tensor_d0 & 0xFFFFu) << 16);                 // tensor_dim0[15:0]
    g1[2] = (int)((tensor_d0 >> 16) | ((tensor_d1 & 0xFFFFu) << 16));
    g1[3] = (int)((tensor_d1 >> 16) | (tile_d0 << 16));         // tile_dim0
    g1[4] = (int)(tile_d1 & 0xFFFFu);                           // tile_dim1 (tile_dim2=0)
    g1[5] = (int)(unsigned int)stride_elems;                    // dim0 stride lo32
    g1[6] = (int)((unsigned int)(stride_elems >> 32) & 0xFFFFu);// stride hi16 (dim1_stride=0)
    g1[7] = 0;
    v4i z = {0, 0, 0, 0};
#ifdef TDM_6ARG
    v8i z8 = {0, 0, 0, 0, 0, 0, 0, 0};
    __builtin_amdgcn_tensor_load_to_lds(g0, g1, z, z, z8, 0);
#else
    __builtin_amdgcn_tensor_load_to_lds(g0, g1, z, z, 0);
#endif
}

// ---------------------------------------------------------------------------
// f32 -> packed bf16 conversion (8 elements / thread)
// ---------------------------------------------------------------------------
__global__ __launch_bounds__(256) void cvt_bf16_kernel(
    const float* __restrict__ in, unsigned int* __restrict__ out)
{
    const size_t i = (size_t)blockIdx.x * 256 + threadIdx.x;
    const float4* in4 = (const float4*)in;
    float4 a = in4[2 * i];
    float4 b = in4[2 * i + 1];
    uint4 o;
    o.x = pack_bf16(a.x, a.y);
    o.y = pack_bf16(a.z, a.w);
    o.z = pack_bf16(b.x, b.y);
    o.w = pack_bf16(b.z, b.w);
    ((uint4*)out)[i] = o;
}

// ---------------------------------------------------------------------------
// GEMM: act(A[M,K]bf16 @ W[N,K]bf16^T + bias[N]) -> optional f32 / bf16 outputs
// 256 thr (8 waves), 128x128 tile, K-step 32, TDM-staged, double-buffered LDS
// ---------------------------------------------------------------------------
#define LDT 17   // dwords per LDS tile row: 16 data (32 bf16) + 1 TDM pad dword

__global__ __launch_bounds__(256) void gemm_bias_act(
    const unsigned short* __restrict__ A, const unsigned short* __restrict__ W,
    const float* __restrict__ bias,
    float* __restrict__ Cf, unsigned short* __restrict__ Cb,
    int M, int N, int K, int relu)
{
    __shared__ unsigned int As[2][128 * LDT];
    __shared__ unsigned int Ws[2][128 * LDT];

    const int tid  = threadIdx.x;
    const int lane = tid & 31;
    const int wave = tid >> 5;
    const int half = lane >> 4;
    const int l16  = lane & 15;

    const int bm = blockIdx.y * 128;
    const int bn = blockIdx.x * 128;
    const int mBase = (wave >> 1) * 32;   // 4 waves along M
    const int nBase = (wave & 1) * 64;    // 2 waves along N

    v8f acc[2][4];
    #pragma unroll
    for (int i = 0; i < 2; ++i)
        #pragma unroll
        for (int j = 0; j < 4; ++j)
            #pragma unroll
            for (int r = 0; r < 8; ++r) acc[i][j][r] = 0.0f;

    const char* Ag = (const char*)(A + (size_t)bm * K);
    const char* Wg = (const char*)(W + (size_t)bn * K);
    const int nK = K >> 5;

    if (wave == 0) {  // prologue: stage k-slab 0 into buffer 0 via TDM
        tdm_load_2d_bf16((unsigned int)(__SIZE_TYPE__)&As[0][0], Ag, K, M, 32, 128, K, 3);
        tdm_load_2d_bf16((unsigned int)(__SIZE_TYPE__)&Ws[0][0], Wg, K, N, 32, 128, K, 3);
    }

    for (int kt = 0; kt < nK; ++kt) {
        const int cur = kt & 1;
        if (wave == 0) __builtin_amdgcn_s_wait_tensorcnt(0);
        __syncthreads();                                 // tile[cur] visible to all waves
        if (wave == 0 && kt + 1 < nK) {                  // prefetch next slab into other buffer
            tdm_load_2d_bf16((unsigned int)(__SIZE_TYPE__)&As[cur ^ 1][0],
                             Ag + (size_t)(kt + 1) * 64, K, M, 32, 128, K, 3);
            tdm_load_2d_bf16((unsigned int)(__SIZE_TYPE__)&Ws[cur ^ 1][0],
                             Wg + (size_t)(kt + 1) * 64, K, N, 32, 128, K, 3);
        }

        v16bf a[2], b[4];
        #pragma unroll
        for (int i = 0; i < 2; ++i) {
            int m = mBase + i * 16 + l16;
            v8u u;
            #pragma unroll
            for (int v = 0; v < 8; ++v) {
                int kd = ((v >> 2) * 8) + half * 4 + (v & 3);   // A 16x32 bf16 layout
                u[v] = As[cur][m * LDT + kd];
            }
            a[i] = as_v16bf(u);
        }
        #pragma unroll
        for (int j = 0; j < 4; ++j) {
            int n = nBase + j * 16 + l16;
            v8u u;
            #pragma unroll
            for (int v = 0; v < 8; ++v) {
                int kd = ((v >> 2) * 8) + half * 4 + (v & 3);   // B 32x16 bf16 layout
                u[v] = Ws[cur][n * LDT + kd];
            }
            b[j] = as_v16bf(u);
        }
        #pragma unroll
        for (int i = 0; i < 2; ++i)
            #pragma unroll
            for (int j = 0; j < 4; ++j)
                acc[i][j] = WMMA_BF16(a[i], b[j], acc[i][j]);
        __syncthreads();                                 // reads done before buffer reuse
    }

    // epilogue: bias + optional ReLU; C layout: lane -> N=l16, VGPR r -> M=r+8*half
    #pragma unroll
    for (int i = 0; i < 2; ++i) {
        #pragma unroll
        for (int j = 0; j < 4; ++j) {
            int col = bn + nBase + j * 16 + l16;
            float bv = bias[col];
            #pragma unroll
            for (int r = 0; r < 8; ++r) {
                int row = bm + mBase + i * 16 + r + 8 * half;
                float val = acc[i][j][r] + bv;
                if (relu) val = fmaxf(val, 0.0f);
                if (Cf) Cf[(size_t)row * N + col] = val;
                if (Cb) Cb[(size_t)row * N + col] = bf16_of(val);
            }
        }
    }
}

// ---------------------------------------------------------------------------
// Flash attention: one block (256 thr, 8 waves) per (b,h). K head staged by
// TDM (1024x64 bf16, 33-dword padded rows); V repacked transposed; Q operands
// loaded as packed bf16 dwords from global. Online softmax over 32-key blocks.
// ---------------------------------------------------------------------------
#define KS_STRIDE 33    // dwords per K row (32 data + 1 TDM pad)
#define VS_STRIDE 514   // dwords per V feature row (512 data + 2 pad)
#define KS_DW (1024 * KS_STRIDE)
#define VS_DW (64 * VS_STRIDE)
#define PS_DW (8 * 16 * 16)
#define ATTN_SMEM_BYTES ((KS_DW + VS_DW + PS_DW) * 4)

__global__ __launch_bounds__(256) void attention_kernel(
    const unsigned short* __restrict__ Q, const unsigned short* __restrict__ K,
    const unsigned short* __restrict__ V, float* __restrict__ Ctx)
{
    extern __shared__ unsigned int smem[];
    unsigned int* Ks = smem;
    unsigned int* Vs = smem + KS_DW;
    unsigned int* Ps = smem + KS_DW + VS_DW;

    const int bh = blockIdx.x;
    const int b  = bh >> 4;
    const int h  = bh & 15;
    const size_t base = (size_t)b * 1024 * 1024 + (size_t)h * 64;  // (b, t=0, h*64)

    const int tid  = threadIdx.x;
    const int lane = tid & 31;
    const int wave = tid >> 5;
    const int half = lane >> 4;
    const int l16  = lane & 15;

    // K head tile via TDM: rows t=0..1023, 64 bf16 each, row stride D=1024 elems
    if (wave == 0)
        tdm_load_2d_bf16((unsigned int)(__SIZE_TYPE__)Ks, (const char*)(K + base),
                         1024, 8192, 64, 1024, 1024, 4);

    // V transposed: Vs[f][td] = pack(V[2td][f], V[2td+1][f]) (already bf16)
    for (int i = 0; i < 128; ++i) {
        int linear = tid + 256 * i;
        int f = linear >> 9, td = linear & 511;
        unsigned int v0 = V[base + (size_t)(2 * td) * 1024 + f];
        unsigned int v1 = V[base + (size_t)(2 * td + 1) * 1024 + f];
        Vs[f * VS_STRIDE + td] = v0 | (v1 << 16);
    }
    if (wave == 0) __builtin_amdgcn_s_wait_tensorcnt(0);
    __syncthreads();

    unsigned int*   Pw = Ps + wave * 256;
    unsigned short* Pb = (unsigned short*)Pw;

    for (int qt = 0; qt < 8; ++qt) {
        const int qRow0 = wave * 128 + qt * 16;

        // Q operand: packed bf16 dwords straight from global (A 16x32 layout)
        v16bf qa[2];
        {
            const unsigned int* Qr =
                (const unsigned int*)(Q + base + (size_t)(qRow0 + l16) * 1024);
            v8u u0, u1;
            #pragma unroll
            for (int v = 0; v < 8; ++v) {
                int kd = ((v >> 2) * 8) + half * 4 + (v & 3);
                u0[v] = Qr[kd];
                u1[v] = Qr[16 + kd];
            }
            qa[0] = as_v16bf(u0); qa[1] = as_v16bf(u1);
        }

        float mrow[8], lrow[8];
        v8f ctx[4];
        #pragma unroll
        for (int r = 0; r < 8; ++r) { mrow[r] = -INFINITY; lrow[r] = 0.0f; }
        #pragma unroll
        for (int j = 0; j < 4; ++j)
            #pragma unroll
            for (int r = 0; r < 8; ++r) ctx[j][r] = 0.0f;

        for (int tb = 0; tb < 32; ++tb) {
            v8f s0, s1;
            #pragma unroll
            for (int r = 0; r < 8; ++r) { s0[r] = 0.0f; s1[r] = 0.0f; }

            #pragma unroll
            for (int th = 0; th < 2; ++th) {
                int t = tb * 32 + th * 16 + l16;
                v8u u0, u1;
                #pragma unroll
                for (int v = 0; v < 8; ++v) {
                    int kd = ((v >> 2) * 8) + half * 4 + (v & 3);
                    u0[v] = Ks[t * KS_STRIDE + kd];
                    u1[v] = Ks[t * KS_STRIDE + 16 + kd];
                }
                v16bf kb0 = as_v16bf(u0), kb1 = as_v16bf(u1);
                if (th == 0) { s0 = WMMA_BF16(qa[0], kb0, s0); s0 = WMMA_BF16(qa[1], kb1, s0); }
                else         { s1 = WMMA_BF16(qa[0], kb0, s1); s1 = WMMA_BF16(qa[1], kb1, s1); }
            }

            // online softmax; 1/sqrt(DH)=0.125 folded here
            float scale[8];
            #pragma unroll
            for (int r = 0; r < 8; ++r) {
                float a0 = s0[r] * 0.125f;
                float a1 = s1[r] * 0.125f;
                float mx = fmaxf(a0, a1);
                #pragma unroll
                for (int msk = 8; msk; msk >>= 1) mx = fmaxf(mx, __shfl_xor(mx, msk, 32));
                float mnew = fmaxf(mrow[r], mx);
                float sc   = __expf(mrow[r] - mnew);
                float p0   = __expf(a0 - mnew);
                float p1   = __expf(a1 - mnew);
                float rs   = p0 + p1;
                #pragma unroll
                for (int msk = 8; msk; msk >>= 1) rs += __shfl_xor(rs, msk, 32);
                lrow[r] = lrow[r] * sc + rs;
                mrow[r] = mnew;
                scale[r] = sc;
                s0[r] = p0; s1[r] = p1;
            }
            #pragma unroll
            for (int j = 0; j < 4; ++j)
                #pragma unroll
                for (int r = 0; r < 8; ++r) ctx[j][r] *= scale[r];

            // re-layout P (C-layout -> A-layout) via per-wave LDS scratch
            #pragma unroll
            for (int r = 0; r < 8; ++r) {
                int m = r + 8 * half;
                Pb[m * 32 + l16]      = bf16_of(s0[r]);
                Pb[m * 32 + 16 + l16] = bf16_of(s1[r]);
            }
            asm volatile("s_wait_dscnt 0x0" ::: "memory");   // same-wave LDS RAW

            v16bf pa;
            {
                v8u u;
                #pragma unroll
                for (int v = 0; v < 8; ++v) {
                    int kd = ((v >> 2) * 8) + half * 4 + (v & 3);
                    u[v] = Pw[l16 * 16 + kd];
                }
                pa = as_v16bf(u);
            }
            #pragma unroll
            for (int j = 0; j < 4; ++j) {
                int f = j * 16 + l16;
                v8u u;
                #pragma unroll
                for (int v = 0; v < 8; ++v) {
                    int kd = ((v >> 2) * 8) + half * 4 + (v & 3);
                    u[v] = Vs[f * VS_STRIDE + tb * 16 + kd];
                }
                ctx[j] = WMMA_BF16(pa, as_v16bf(u), ctx[j]);
            }
            asm volatile("s_wait_dscnt 0x0" ::: "memory");   // WAR before P overwrite
        }

        #pragma unroll
        for (int r = 0; r < 8; ++r) {
            float inv = 1.0f / lrow[r];
            int m = qRow0 + r + 8 * half;
            #pragma unroll
            for (int j = 0; j < 4; ++j)
                Ctx[base + (size_t)m * 1024 + j * 16 + l16] = ctx[j][r] * inv;
        }
    }
}

// ---------------------------------------------------------------------------
// residual add + LayerNorm -> optional f32 / bf16 outputs; one block per row
// ---------------------------------------------------------------------------
__global__ __launch_bounds__(256) void add_ln_kernel(
    const float* __restrict__ x, const float* __restrict__ res,
    const float* __restrict__ g, const float* __restrict__ bb,
    float* __restrict__ outf, unsigned short* __restrict__ outb)
{
    const int row = blockIdx.x;
    const float* xr = x   + (size_t)row * 1024;
    const float* rr = res + (size_t)row * 1024;
    const int tid = threadIdx.x, lane = tid & 31, wave = tid >> 5;

    float vals[4];
    float s1 = 0.0f, s2 = 0.0f;
    #pragma unroll
    for (int i = 0; i < 4; ++i) {
        int c = tid + 256 * i;
        float v = xr[c] + rr[c];
        vals[i] = v; s1 += v; s2 += v * v;
    }
    #pragma unroll
    for (int msk = 16; msk; msk >>= 1) {
        s1 += __shfl_xor(s1, msk, 32);
        s2 += __shfl_xor(s2, msk, 32);
    }
    __shared__ float red[2][8];
    if (lane == 0) { red[0][wave] = s1; red[1][wave] = s2; }
    __syncthreads();
    float t1 = 0.0f, t2 = 0.0f;
    #pragma unroll
    for (int w = 0; w < 8; ++w) { t1 += red[0][w]; t2 += red[1][w]; }
    float mu   = t1 * (1.0f / 1024.0f);
    float var  = t2 * (1.0f / 1024.0f) - mu * mu;
    float rstd = rsqrtf(var + 1e-8f);
    #pragma unroll
    for (int i = 0; i < 4; ++i) {
        int c = tid + 256 * i;
        float o = (vals[i] - mu) * rstd * g[c] + bb[c];
        if (outf) outf[(size_t)row * 1024 + c] = o;
        if (outb) outb[(size_t)row * 1024 + c] = bf16_of(o);
    }
}

// ---------------------------------------------------------------------------
// launch
// ---------------------------------------------------------------------------
extern "C" void kernel_launch(void* const* d_in, const int* in_sizes, int n_in,
                              void* d_out, int out_size, void* d_ws, size_t ws_size,
                              hipStream_t stream) {
    (void)in_sizes; (void)n_in; (void)out_size; (void)ws_size;
    const float* q     = (const float*)d_in[0];
    const float* Qw    = (const float*)d_in[1];
    const float* Qbias = (const float*)d_in[2];
    const float* Kw    = (const float*)d_in[3];
    const float* Kbias = (const float*)d_in[4];
    const float* Vw    = (const float*)d_in[5];
    const float* Vbias = (const float*)d_in[6];
    const float* ln_g  = (const float*)d_in[7];
    const float* ln_b  = (const float*)d_in[8];
    const float* ff1w  = (const float*)d_in[9];
    const float* ff1b  = (const float*)d_in[10];
    const float* ff2w  = (const float*)d_in[11];
    const float* ff2b  = (const float*)d_in[12];
    const float* flng  = (const float*)d_in[13];
    const float* flnb  = (const float*)d_in[14];
    const float* pw    = (const float*)d_in[15];
    const float* pb    = (const float*)d_in[16];

    // workspace carve-up (all sizes 256B-aligned powers of two)
    char* p = (char*)d_ws;
    auto alloc = [&](size_t bytes) { char* r = p; p += bytes; return r; };
    const size_t NE  = (size_t)8192 * 1024;        // (B*S, D) elements
    unsigned short* qbf   = (unsigned short*)alloc(NE * 2);
    unsigned short* QwB   = (unsigned short*)alloc((size_t)1024 * 1024 * 2);
    unsigned short* KwB   = (unsigned short*)alloc((size_t)1024 * 1024 * 2);
    unsigned short* VwB   = (unsigned short*)alloc((size_t)1024 * 1024 * 2);
    unsigned short* F1B   = (unsigned short*)alloc((size_t)4096 * 1024 * 2);
    unsigned short* F2B   = (unsigned short*)alloc((size_t)1024 * 4096 * 2);
    unsigned short* PwB   = (unsigned short*)alloc((size_t)1024 * 1024 * 2);
    unsigned short* Qbf   = (unsigned short*)alloc(NE * 2);
    unsigned short* Kbf   = (unsigned short*)alloc(NE * 2);
    unsigned short* Vbf   = (unsigned short*)alloc(NE * 2);
    float*          Ctx   = (float*)alloc(NE * 4);
    float*          Out1f = (float*)alloc(NE * 4);
    unsigned short* Out1b = (unsigned short*)alloc(NE * 2);
    unsigned short* Hbf   = (unsigned short*)alloc((size_t)8192 * 4096 * 2);
    float*          FFb   = (float*)alloc(NE * 4);
    unsigned short* Out2b = (unsigned short*)alloc(NE * 2);

    dim3 blk(256);
    // one-time f32 -> bf16 conversions (8 f32 per thread)
    cvt_bf16_kernel<<<4096, blk, 0, stream>>>(q,    (unsigned int*)qbf);
    cvt_bf16_kernel<<<512,  blk, 0, stream>>>(Qw,   (unsigned int*)QwB);
    cvt_bf16_kernel<<<512,  blk, 0, stream>>>(Kw,   (unsigned int*)KwB);
    cvt_bf16_kernel<<<512,  blk, 0, stream>>>(Vw,   (unsigned int*)VwB);
    cvt_bf16_kernel<<<2048, blk, 0, stream>>>(ff1w, (unsigned int*)F1B);
    cvt_bf16_kernel<<<2048, blk, 0, stream>>>(ff2w, (unsigned int*)F2B);
    cvt_bf16_kernel<<<512,  blk, 0, stream>>>(pw,   (unsigned int*)PwB);

    // QKV projections (relu) -> bf16 heads
    gemm_bias_act<<<dim3(8, 64), blk, 0, stream>>>(qbf, QwB, Qbias, nullptr, Qbf, 8192, 1024, 1024, 1);
    gemm_bias_act<<<dim3(8, 64), blk, 0, stream>>>(qbf, KwB, Kbias, nullptr, Kbf, 8192, 1024, 1024, 1);
    gemm_bias_act<<<dim3(8, 64), blk, 0, stream>>>(qbf, VwB, Vbias, nullptr, Vbf, 8192, 1024, 1024, 1);

    // attention (128 blocks = B*H), ~275 KB dynamic LDS
    (void)hipFuncSetAttribute(reinterpret_cast<const void*>(attention_kernel),
                              hipFuncAttributeMaxDynamicSharedMemorySize,
                              (int)ATTN_SMEM_BYTES);
    attention_kernel<<<128, blk, ATTN_SMEM_BYTES, stream>>>(Qbf, Kbf, Vbf, Ctx);

    add_ln_kernel<<<8192, blk, 0, stream>>>(Ctx, q, ln_g, ln_b, Out1f, Out1b);

    gemm_bias_act<<<dim3(32, 64), blk, 0, stream>>>(Out1b, F1B, ff1b, nullptr, Hbf, 8192, 4096, 1024, 1);
    gemm_bias_act<<<dim3(8, 64),  blk, 0, stream>>>(Hbf, F2B, ff2b, FFb, nullptr, 8192, 1024, 4096, 0);

    add_ln_kernel<<<8192, blk, 0, stream>>>(FFb, Out1f, flng, flnb, nullptr, Out2b);

    gemm_bias_act<<<dim3(8, 64), blk, 0, stream>>>(Out2b, PwB, pb, (float*)d_out, nullptr, 8192, 1024, 1024, 0);
}
